// DNCCell_6081673691270
// MI455X (gfx1250) — matrix operand order repaired and checked by reading, not compiled, not run on hardware
//
#include <hip/hip_runtime.h>
#include <math.h>

typedef __attribute__((ext_vector_type(16))) _Float16 v16h;
typedef __attribute__((ext_vector_type(8)))  _Float16 v8h;
typedef __attribute__((ext_vector_type(8)))  float    v8f;

#define B_  64
#define T_  256
#define U_  512
#define M_  16
#define BT  8      // batch rows per workgroup
#define RHP 520    // padded row stride (halves) for read_h

// One-shot prep: WtT[n][k] = (f16) Wt[k][n]
__global__ void wt_transpose_f16(const float* __restrict__ Wt, _Float16* __restrict__ WtT) {
    int idx = blockIdx.x * blockDim.x + threadIdx.x;   // n*U + k
    int n = idx >> 9, k = idx & (U_ - 1);
    WtT[idx] = (_Float16)Wt[(size_t)k * U_ + n];
}

// WGP-scope prefetch (scope field omitted == 0 == WGP: pulls into all cache
// levels; __builtin_prefetch only reaches SE/SYS which stop at GL2).
// - immediate offset: no per-prefetch address VALU, no WAR hazard on the
//   wmma operand registers
// - no "memory" clobber: a prefetch is a hint; the clobber would act as a
//   scheduling fence and destroy load/wmma software pipelining
__device__ __forceinline__ void prefetch_wgp(const void* p, int off) {
    asm volatile("global_prefetch_b8 %0, off offset:%c1" :: "v"(p), "i"(off));
}

__global__ __launch_bounds__(256, 1)
void dnc_scan(const float* __restrict__ x,   const float* __restrict__ mem0,
              const float* __restrict__ Wr,  const float* __restrict__ br,
              const _Float16* __restrict__ WtT, const float* __restrict__ bt,
              const float* __restrict__ Ww,  const float* __restrict__ bw,
              float* __restrict__ out) {
    // A operand for WMMA (8 real batch rows + 8 zero rows), f16
    __shared__ __attribute__((aligned(16))) _Float16 read_h[16][RHP];
    __shared__ __attribute__((aligned(16))) float    tr_s[BT][U_];
    // step-invariant small operands, staged in LDS once
    __shared__ float WrS[U_][17];
    __shared__ float WwS[U_][17];   // padded 16 -> 17 dwords/row: conflict-free
    __shared__ float btS[U_];
    __shared__ float brS[17];
    __shared__ float bwS[16];

    const int tid  = threadIdx.x;
    const int wave = tid >> 5;                 // local batch row in elementwise phases
    const int lane = tid & 31;
    const int bg   = blockIdx.x * BT + wave;   // global batch row
    const int lhi  = lane >> 4;
    const int l16  = lane & 15;

    // thread->u mapping: u = lane + 32*i  (lane-consecutive at fixed i:
    //  conflict-free LDS reads, 128B-coalesced global access per i)
    // ---- recurrent state in registers: mem[m][i] = mem[bg][m][lane+32i] ----
    float mem[M_][16];
    #pragma unroll
    for (int m = 0; m < M_; ++m) {
        const float* mp = mem0 + (size_t)bg * (M_ * U_) + (size_t)m * U_ + lane;
        #pragma unroll
        for (int i = 0; i < 16; ++i) mem[m][i] = mp[32 * i];
    }

    // one-time LDS staging
    for (int idx = tid; idx < U_ * 17; idx += 256) ((float*)WrS)[idx] = Wr[idx];
    for (int idx = tid; idx < U_ * 16; idx += 256) WwS[idx >> 4][idx & 15] = Ww[idx];
    for (int idx = tid; idx < U_; idx += 256)      btS[idx] = bt[idx];
    if (tid < 17) brS[tid] = br[tid];
    if (tid < 16) bwS[tid] = bw[tid];
    for (int idx = tid; idx < 8 * RHP; idx += 256)           // zero fake A rows 8..15
        read_h[8 + idx / RHP][idx % RHP] = (_Float16)0.0f;
    __syncthreads();

    // base of this wave's 4 N-tile rows in WtT (immediate offsets cover the rest)
    const _Float16* bbase = WtT + (size_t)(wave * 64 + l16) * U_ + lhi * 16;

    for (int t = 0; t < T_; ++t) {
        const float* xp = x + ((size_t)bg * T_ + t) * U_ + lane;

        // ---------------- Phase 1: att-mean + Wr logits + softmax (fp32) ----------
        float part[17];
        #pragma unroll
        for (int k = 0; k < 17; ++k) part[k] = 0.0f;
        #pragma unroll
        for (int i = 0; i < 16; ++i) {
            float s = xp[32 * i];
            #pragma unroll
            for (int m = 0; m < M_; ++m) s += mem[m][i];
            float am = s * (1.0f / 17.0f);
            const float* wr = &WrS[lane + 32 * i][0];
            #pragma unroll
            for (int k = 0; k < 17; ++k) part[k] = fmaf(am, wr[k], part[k]);
        }
        #pragma unroll
        for (int k = 0; k < 17; ++k) {           // wave32 butterfly over u
            float v = part[k];
            v += __shfl_xor(v, 16, 32);
            v += __shfl_xor(v, 8, 32);
            v += __shfl_xor(v, 4, 32);
            v += __shfl_xor(v, 2, 32);
            v += __shfl_xor(v, 1, 32);
            part[k] = v;
        }
        float lg[17], mx;
        lg[0] = part[0] + brS[0]; mx = lg[0];
        #pragma unroll
        for (int k = 1; k < 17; ++k) { lg[k] = part[k] + brS[k]; mx = fmaxf(mx, lg[k]); }
        float rw[17], ssum = 0.0f;
        #pragma unroll
        for (int k = 0; k < 17; ++k) { rw[k] = expf(lg[k] - mx); ssum += rw[k]; }
        float inv = 1.0f / ssum;
        #pragma unroll
        for (int k = 0; k < 17; ++k) rw[k] *= inv;

        // ---------------- Phase 2: weighted read (thread-local mem) -> f16 A ------
        #pragma unroll
        for (int i = 0; i < 16; ++i) {
            float rv = rw[16] * xp[32 * i];
            #pragma unroll
            for (int m = 0; m < M_; ++m) rv = fmaf(rw[m], mem[m][i], rv);
            read_h[wave][lane + 32 * i] = (_Float16)rv;
        }
        __syncthreads();

        // ---------------- Phase 3: tr = relu(read @ Wt + bt) via WMMA f16 ---------
        v8f acc[4];
        #pragma unroll
        for (int tt = 0; tt < 4; ++tt)
            #pragma unroll
            for (int r = 0; r < 8; ++r) acc[tt][r] = 0.0f;

        const int abase = lhi * 16;
        #pragma unroll
        for (int kb = 0; kb < 16; ++kb) {        // fully unrolled: 64 static wmma
            v16h a;
            const _Float16* ap = &read_h[l16][kb * 32 + abase];
            ((v8h*)&a)[0] = *(const v8h*)(ap);
            ((v8h*)&a)[1] = *(const v8h*)(ap + 8);
            #pragma unroll
            for (int tt = 0; tt < 4; ++tt) {
                const _Float16* bp = bbase + (size_t)tt * 16 * U_ + kb * 32;
                v16h bm;
                ((v8h*)&bm)[0] = *(const v8h*)(bp);
                ((v8h*)&bm)[1] = *(const v8h*)(bp + 8);
                acc[tt] = __builtin_amdgcn_wmma_f32_16x16x32_f16(
                    false, a, false, bm, (short)0, acc[tt], false, false);
            }
            // one WGP-scope prefetch per 128B line, 2 k-blocks ahead (in-flight
            // window ~64KB across the WG == WGP$ capacity). Compile-time guards:
            // never past the 512KB WtT buffer (WGP prefetch is non-speculative).
            if (((kb & 1) == 0) && kb < 14) {
                #pragma unroll
                for (int tt = 0; tt < 4; ++tt)
                    prefetch_wgp(bbase, tt * 16384 + (kb + 2) * 64);
            }
        }
        if (lhi == 0) {   // D rows 0..7 (real batch rows) live in lanes 0-15
            #pragma unroll
            for (int tt = 0; tt < 4; ++tt) {
                int n = wave * 64 + tt * 16 + l16;
                float btv = btS[n];
                #pragma unroll
                for (int r = 0; r < BT; ++r) {
                    float v = acc[tt][r] + btv;
                    tr_s[r][n] = v > 0.0f ? v : 0.0f;
                }
            }
        }
        __syncthreads();

        // ---------------- Phase 4: ww = sigmoid(tr @ Ww + bw) ---------------------
        float wp[16];
        #pragma unroll
        for (int k = 0; k < 16; ++k) wp[k] = 0.0f;
        #pragma unroll
        for (int i = 0; i < 16; ++i) {
            float trv = tr_s[wave][lane + 32 * i];
            const float* wwp = &WwS[lane + 32 * i][0];
            #pragma unroll
            for (int k = 0; k < 16; ++k) wp[k] = fmaf(trv, wwp[k], wp[k]);
        }
        #pragma unroll
        for (int k = 0; k < 16; ++k) {
            float v = wp[k];
            v += __shfl_xor(v, 16, 32);
            v += __shfl_xor(v, 8, 32);
            v += __shfl_xor(v, 4, 32);
            v += __shfl_xor(v, 2, 32);
            v += __shfl_xor(v, 1, 32);
            wp[k] = v;
        }
        float wwv[16];
        #pragma unroll
        for (int k = 0; k < 16; ++k)
            wwv[k] = 1.0f / (1.0f + expf(-(wp[k] + bwS[k])));

        // ---------------- Phase 5: mem convex update + y store --------------------
        float* op = out + ((size_t)bg * T_ + t) * U_ + lane;
        #pragma unroll
        for (int i = 0; i < 16; ++i) {
            float trv = tr_s[wave][lane + 32 * i];
            op[32 * i] = xp[32 * i] + trv;        // x reload hits the hot L2/L0 line
            #pragma unroll
            for (int m = 0; m < M_; ++m)
                mem[m][i] = fmaf(wwv[m], trv - mem[m][i], mem[m][i]);
        }
        // tr_s is only re-written after next step's post-Phase-2 barrier, which
        // cannot be passed until every wave finishes Phase 5 here.
    }
}

extern "C" void kernel_launch(void* const* d_in, const int* in_sizes, int n_in,
                              void* d_out, int out_size, void* d_ws, size_t ws_size,
                              hipStream_t stream) {
    const float* xin  = (const float*)d_in[0];
    const float* mem0 = (const float*)d_in[1];
    const float* Wr   = (const float*)d_in[2];
    const float* br   = (const float*)d_in[3];
    const float* Wt   = (const float*)d_in[4];
    const float* bt   = (const float*)d_in[5];
    const float* Ww   = (const float*)d_in[6];
    const float* bw   = (const float*)d_in[7];
    float* out        = (float*)d_out;

    _Float16* WtT = (_Float16*)d_ws;   // 512*512*2 = 512 KB scratch

    wt_transpose_f16<<<(U_ * U_) / 256, 256, 0, stream>>>(Wt, WtT);
    dnc_scan<<<B_ / BT, 256, 0, stream>>>(xin, mem0, Wr, br, WtT, bt, Ww, bw, out);
}
